// LiteFlowHead_11218454577863
// MI455X (gfx1250) — matrix operand order repaired and checked by reading, not compiled, not run on hardware
//
#include <hip/hip_runtime.h>
#include <hip/hip_bf16.h>
#include <math.h>

// ---------------------------------------------------------------------------
// LiteFlowHead on MI455X (gfx1250). All channel contractions via
// v_wmma_f32_16x16x32_f16 with fp16 NHWC activations (channel-padded to 32)
// and pre-packed fp16 weights. Each wave register-blocks a 32x32 output tile
// (4 WMMAs per K-step from 4 fragment loads -> 2x arithmetic intensity vs 1x1).
// The 3x3 up-mask conv runs 9 taps fused in-register over a zero-padded input.
// ---------------------------------------------------------------------------

typedef __attribute__((ext_vector_type(16))) _Float16 v16h;
typedef __attribute__((ext_vector_type(8)))  _Float16 v8h;
typedef __attribute__((ext_vector_type(8)))  float    v8f;

#define EF_BIAS   1
#define EF_BN     2
#define EF_SILU   4
#define EF_RELU   8
#define EF_OUTF32 16   // store float32 (else fp16 NHWC)
#define EF_CHW    32   // f32 output in CHW [O,S] layout (per batch)
#define EF_ACC    64   // preload accumulator from f32 CHW OUT
#define EF_CONV3  128  // 9-tap 3x3 conv over spatially padded input

__device__ __forceinline__ float silu_f(float x) { return x / (1.0f + __expf(-x)); }

// A 16x32 fragment: lane l,g: halves i -> K=g*8+i ; halves 8+i -> K=16+g*8+i
__device__ __forceinline__ v16h load_afrag(const _Float16* p, int g) {
    v8h lo = *(const v8h*)(p + g * 8);
    v8h hi = *(const v8h*)(p + 16 + g * 8);
    return __builtin_shufflevector(lo, hi, 0,1,2,3,4,5,6,7,8,9,10,11,12,13,14,15);
}
// B 32x16 fragment: lane l,g: halves i -> K=g*8+i ; halves 8+i -> K=16+g*8+i
__device__ __forceinline__ v16h load_bfrag(const _Float16* p, int g) {
    v8h lo = *(const v8h*)(p + g * 8);
    v8h hi = *(const v8h*)(p + 16 + g * 8);
    return __builtin_shufflevector(lo, hi, 0,1,2,3,4,5,6,7,8,9,10,11,12,13,14,15);
}

// ---------------------------------------------------------------------------
// WMMA GEMM, M = spatial, N = out channel, K = in channel.
// Wave tile: 32(M) x 32(N), accumulators acc[mi*2+ni].
//   A: fp16 NHWC activations [.., row, Kpad] (CONV3: rows in padded (H+2)x(W+2))
//   B: fp16 packed weights   [ntaps][Opad][Kpad]
// C/D layout: lane l,g: elem r -> (M = g*8+r, N = l) within each 16x16 subtile.
// ---------------------------------------------------------------------------
__global__ __launch_bounds__(128)
void gemm16_kernel(const _Float16* __restrict__ A, const _Float16* __restrict__ Bw,
                   void* __restrict__ OUT,
                   const float* __restrict__ bias,
                   const float* __restrict__ bng, const float* __restrict__ bnb,
                   const float* __restrict__ bnm, const float* __restrict__ bnv,
                   int O, int Opad, int Kpad, int H, int W,
                   long long aBS, long long oBS, int ldo, int flags)
{
    const int S = H * W;
    const int lane = threadIdx.x & 31;
    const int wave = threadIdx.x >> 5;
    const int s0 = blockIdx.x * 32;
    const int o0 = blockIdx.y * 128 + wave * 32;
    if (o0 >= Opad) return;                       // wave-uniform
    const int g = lane >> 4, l = lane & 15;

    const _Float16* Ab = A + (long long)blockIdx.z * aBS;
    const bool conv3 = (flags & EF_CONV3) != 0;
    const int Wp = W + 2;

    // two A row sets per lane (M subtiles)
    long long arow0, arow1;
    {
        int sA0 = s0 + l, sA1 = s0 + 16 + l;
        if (conv3) {
            int h0 = sA0 / W, w0 = sA0 - h0 * W;
            int h1 = sA1 / W, w1 = sA1 - h1 * W;
            arow0 = (long long)(h0 * Wp + w0) * Kpad;
            arow1 = (long long)(h1 * Wp + w1) * Kpad;
        } else {
            arow0 = (long long)sA0 * Kpad;
            arow1 = (long long)sA1 * Kpad;
        }
    }
    const _Float16* Aptr0 = Ab + arow0;
    const _Float16* Aptr1 = Ab + arow1;
    // two B row sets per lane (N subtiles)
    const long long brow0 = (long long)(o0 + l) * Kpad;
    const long long brow1 = (long long)(o0 + 16 + l) * Kpad;

    v8f acc[4] = {v8f{}, v8f{}, v8f{}, v8f{}};
    if (flags & EF_ACC) {                          // f32 CHW accumulate (flow head)
        const float* outF = (const float*)OUT + (long long)blockIdx.z * oBS;
#pragma unroll
        for (int ni = 0; ni < 2; ++ni) {
            int o = o0 + ni * 16 + l;
            int oc = (o < O) ? o : 0;
#pragma unroll
            for (int mi = 0; mi < 2; ++mi)
#pragma unroll
                for (int r = 0; r < 8; ++r)
                    acc[mi * 2 + ni][r] = outF[(long long)oc * S + (s0 + mi * 16 + g * 8 + r)];
        }
    }

    const int ntaps = conv3 ? 9 : 1;
    for (int t = 0; t < ntaps; ++t) {
        const long long toff = conv3 ? (long long)((t / 3) * Wp + (t % 3)) * Kpad : 0;
        const _Float16* Ap0 = Aptr0 + toff;
        const _Float16* Ap1 = Aptr1 + toff;
        const _Float16* Bp0 = Bw + (long long)t * Opad * Kpad + brow0;
        const _Float16* Bp1 = Bw + (long long)t * Opad * Kpad + brow1;
        for (int k0 = 0; k0 < Kpad; k0 += 32) {
            v16h a0 = load_afrag(Ap0 + k0, g);
            v16h a1 = load_afrag(Ap1 + k0, g);
            v16h b0 = load_bfrag(Bp0 + k0, g);
            v16h b1 = load_bfrag(Bp1 + k0, g);
            acc[0] = __builtin_amdgcn_wmma_f32_16x16x32_f16(false, a0, false, b0, (short)0, acc[0], false, false);
            acc[1] = __builtin_amdgcn_wmma_f32_16x16x32_f16(false, a0, false, b1, (short)0, acc[1], false, false);
            acc[2] = __builtin_amdgcn_wmma_f32_16x16x32_f16(false, a1, false, b0, (short)0, acc[2], false, false);
            acc[3] = __builtin_amdgcn_wmma_f32_16x16x32_f16(false, a1, false, b1, (short)0, acc[3], false, false);
        }
    }

#pragma unroll
    for (int ni = 0; ni < 2; ++ni) {
        int o = o0 + ni * 16 + l;
        int oc = (o < O) ? o : 0;
        float bs = (flags & EF_BIAS) ? bias[oc] : 0.0f;
        float sc = 1.0f, sh = 0.0f;
        if (flags & EF_BN) {
            sc = bng[oc] * rsqrtf(bnv[oc] + 1e-5f);
            sh = bnb[oc] - bnm[oc] * sc;
        }
        if (flags & EF_OUTF32) {
            float* outF = (float*)OUT + (long long)blockIdx.z * oBS;
#pragma unroll
            for (int mi = 0; mi < 2; ++mi)
#pragma unroll
                for (int r = 0; r < 8; ++r) {
                    float v = (acc[mi * 2 + ni][r] + bs) * sc + sh;
                    if (flags & EF_SILU) v = silu_f(v);
                    if (flags & EF_RELU) v = fmaxf(v, 0.0f);
                    int row = s0 + mi * 16 + g * 8 + r;
                    if (o < O) {
                        if (flags & EF_CHW) outF[(long long)o * S + row] = v;
                        else                outF[(long long)row * ldo + o] = v;
                    }
                }
        } else {
            _Float16* out16 = (_Float16*)OUT + (long long)blockIdx.z * oBS;
#pragma unroll
            for (int mi = 0; mi < 2; ++mi)
#pragma unroll
                for (int r = 0; r < 8; ++r) {
                    float v = (acc[mi * 2 + ni][r] + bs) * sc + sh;
                    if (flags & EF_SILU) v = silu_f(v);
                    if (flags & EF_RELU) v = fmaxf(v, 0.0f);
                    int row = s0 + mi * 16 + g * 8 + r;
                    if (o < O) out16[(long long)row * ldo + o] = (_Float16)v;
                }
        }
    }
}

// ---------------- weight pack: f32 [O,K] (strided) -> f16 [Opad,Kpad] --------
__global__ void wprep_kernel(const float* __restrict__ src, _Float16* __restrict__ dst,
                             int O, int K, int Kpad, long long total, int sM, int sK)
{
    long long idx = (long long)blockIdx.x * blockDim.x + threadIdx.x;
    if (idx >= total) return;
    int k = (int)(idx % Kpad); int o = (int)(idx / Kpad);
    float v = (o < O && k < K) ? src[(long long)o * sM + (long long)k * sK] : 0.0f;
    dst[idx] = (_Float16)v;
}

// ---------------- NCHW f32 -> NHWC f16 ---------------------------------------
__global__ void nchw2nhwc_kernel(const float* __restrict__ src, _Float16* __restrict__ dst,
                                 int B, int C, int S)
{
    long long total = (long long)B * S * C;
    long long idx = (long long)blockIdx.x * blockDim.x + threadIdx.x;
    if (idx >= total) return;
    int c = (int)(idx % C); long long t = idx / C;
    int s = (int)(t % S); int b = (int)(t / S);
    dst[idx] = (_Float16)src[((long long)b * C + c) * S + s];
}

// ---------------- depthwise 3x3 (SAME), NHWC f16 -----------------------------
__global__ void dwconv_nhwc_kernel(const _Float16* __restrict__ in, const float* __restrict__ wdw,
                                   _Float16* __restrict__ out, int B, int C, int Kpad,
                                   int H, int W)
{
    int S = H * W;
    long long total = (long long)B * S * Kpad;
    long long idx = (long long)blockIdx.x * blockDim.x + threadIdx.x;
    if (idx >= total) return;
    int c = (int)(idx % Kpad); long long t = idx / Kpad;
    int s = (int)(t % S); int b = (int)(t / S);
    if (c >= C) { out[idx] = (_Float16)0.0f; return; }
    int h = s / W, w = s - h * W;
    const _Float16* ip = in + (long long)b * S * Kpad + c;
    const float* wp = wdw + c * 9;
    float acc = 0.0f;
#pragma unroll
    for (int ky = 0; ky < 3; ++ky) {
        int hh = h + ky - 1; if (hh < 0 || hh >= H) continue;
#pragma unroll
        for (int kx = 0; kx < 3; ++kx) {
            int ww = w + kx - 1; if (ww < 0 || ww >= W) continue;
            acc += (float)ip[(long long)(hh * W + ww) * Kpad] * wp[ky * 3 + kx];
        }
    }
    out[idx] = (_Float16)acc;
}

// ---------------- local correlation (r=4), NHWC f16 --------------------------
__global__ void corr_nhwc_kernel(const _Float16* __restrict__ f1, const _Float16* __restrict__ f2,
                                 _Float16* __restrict__ x0, int B, int C, int H, int W,
                                 int ld0, int corrOff)
{
    int S = H * W;
    long long total = (long long)B * 81 * S;
    long long idx = (long long)blockIdx.x * blockDim.x + threadIdx.x;
    if (idx >= total) return;
    int s = (int)(idx % S); long long t = idx / S;
    int j = (int)(t % 81); int b = (int)(t / 81);
    int dy = j / 9 - 4, dx = j % 9 - 4;
    int h = s / W, w = s - h * W;
    int hh = h + dy, ww = w + dx;
    float acc = 0.0f;
    if (hh >= 0 && hh < H && ww >= 0 && ww < W) {
        const _Float16* p1 = f1 + ((long long)b * S + s) * C;
        const _Float16* p2 = f2 + ((long long)b * S + hh * W + ww) * C;
        for (int c = 0; c < C; ++c) acc += (float)p1[c] * (float)p2[c];
    }
    x0[((long long)b * S + s) * ld0 + corrOff + j] = (_Float16)(acc * rsqrtf((float)C));
}

// ---------------- concat [f1,f2,|f1-f2|,(corr),zero-pad] NHWC ----------------
__global__ void concat3_nhwc_kernel(const _Float16* __restrict__ f1, const _Float16* __restrict__ f2,
                                    _Float16* __restrict__ x0, int B, int P, int S,
                                    int ld0, int corrOff, int corrEnd)
{
    long long total = (long long)B * S * ld0;
    long long idx = (long long)blockIdx.x * blockDim.x + threadIdx.x;
    if (idx >= total) return;
    int c = (int)(idx % ld0); long long t = idx / ld0;
    int s = (int)(t % S); int b = (int)(t / S);
    if (c >= corrOff && c < corrEnd) return;       // written by corr kernel
    long long base = ((long long)b * S + s) * P;
    _Float16 v;
    if (c < P)            v = f1[base + c];
    else if (c < 2 * P)   v = f2[base + (c - P)];
    else if (c < 3 * P) { float a = (float)f1[base + c - 2 * P], bb = (float)f2[base + c - 2 * P];
                          v = (_Float16)fabsf(a - bb); }
    else                  v = (_Float16)0.0f;      // channel pad
    x0[idx] = v;
}

// ---------------- concat [x (f16 NHWC), flow (f32 CHW)] -> f16 NHWC ----------
__global__ void concat_xf_kernel(const _Float16* __restrict__ x, const float* __restrict__ flow,
                                 _Float16* __restrict__ dst, int B, int Cx, int ld, int S)
{
    long long total = (long long)B * S * ld;
    long long idx = (long long)blockIdx.x * blockDim.x + threadIdx.x;
    if (idx >= total) return;
    int c = (int)(idx % ld); long long t = idx / ld;
    int s = (int)(t % S); int b = (int)(t / S);
    _Float16 v;
    if (c < Cx)          v = x[((long long)b * S + s) * Cx + c];
    else if (c < Cx + 2) v = (_Float16)flow[((long long)b * 2 + (c - Cx)) * S + s];
    else                 v = (_Float16)0.0f;
    dst[idx] = v;
}

// ---------------- spatial zero-pad copy (H,W)->(H+2,W+2), NHWC f16 -----------
__global__ void padcpy_kernel(const _Float16* __restrict__ x, _Float16* __restrict__ xp,
                              int B, int C, int H, int W)
{
    int Wp = W + 2, SP = (H + 2) * Wp, S = H * W;
    long long total = (long long)B * SP * C;
    long long idx = (long long)blockIdx.x * blockDim.x + threadIdx.x;
    if (idx >= total) return;
    int c = (int)(idx % C); long long t = idx / C;
    int ps = (int)(t % SP); int b = (int)(t / SP);
    int hp = ps / Wp, wp = ps - hp * Wp;
    int h = hp - 1, w = wp - 1;
    _Float16 v = (h >= 0 && h < H && w >= 0 && w < W)
                     ? x[((long long)b * S + h * W + w) * C + c] : (_Float16)0.0f;
    xp[idx] = v;
}

// ---------------- SE: fused mean + FC1(silu) + FC2(sigmoid) ------------------
__global__ void se_kernel(const _Float16* __restrict__ x,
                          const float* __restrict__ w1, const float* __restrict__ b1,
                          const float* __restrict__ w2, const float* __restrict__ b2,
                          float* __restrict__ scale, int S, int C, int Hd)
{
    int b = blockIdx.x, t = threadIdx.x;           // blockDim == C == 256
    const _Float16* p = x + (long long)b * S * C + t;
    float acc = 0.0f;
    for (int s = 0; s < S; ++s) acc += (float)p[(long long)s * C];
    __shared__ float ssh[256];
    __shared__ float hsh[64];
    ssh[t] = acc / (float)S;
    __syncthreads();
    if (t < Hd) {
        float a = b1[t];
        for (int c = 0; c < C; ++c) a += w1[t * C + c] * ssh[c];
        hsh[t] = silu_f(a);
    }
    __syncthreads();
    float a = b2[t];
    for (int j = 0; j < Hd; ++j) a += w2[t * Hd + j] * hsh[j];
    scale[(long long)b * C + t] = 1.0f / (1.0f + __expf(-a));
}

__global__ void se_scale_kernel(_Float16* __restrict__ x, const float* __restrict__ scale,
                                int C, int S, long long total)
{
    long long idx = (long long)blockIdx.x * blockDim.x + threadIdx.x;
    if (idx >= total) return;
    int c = (int)(idx % C);
    int b = (int)(idx / ((long long)S * C));
    x[idx] = (_Float16)((float)x[idx] * scale[b * C + c]);
}

// ---------------- softmax convex upsample (per batch) ------------------------
__global__ void upsample_kernel(const float* __restrict__ maskb, const float* __restrict__ flowb,
                                float* __restrict__ out, int b, int H, int W, int up)
{
    int S = H * W, UU = up * up, ldm = 9 * UU;
    long long total = (long long)S * UU;
    long long idx = (long long)blockIdx.x * blockDim.x + threadIdx.x;
    if (idx >= total) return;
    int uv = (int)(idx % UU); int s = (int)(idx / UU);
    int u = uv / up, v = uv % up;
    int h = s / W, w = s - h * W;
    float m[9]; float mx = -1e30f;
#pragma unroll
    for (int j = 0; j < 9; ++j) {
        m[j] = maskb[(long long)s * ldm + j * UU + uv];
        mx = fmaxf(mx, m[j]);
    }
    float sum = 0.0f;
#pragma unroll
    for (int j = 0; j < 9; ++j) { m[j] = __expf(m[j] - mx); sum += m[j]; }
    float inv = (float)up / sum;
    int HO = H * up, WO = W * up;
#pragma unroll
    for (int c = 0; c < 2; ++c) {
        float acc = 0.0f;
#pragma unroll
        for (int j = 0; j < 9; ++j) {
            int dy = j / 3 - 1, dx = j % 3 - 1;
            int hh = h + dy, ww = w + dx;
            float fv = (hh >= 0 && hh < H && ww >= 0 && ww < W)
                           ? flowb[(long long)c * S + hh * W + ww] : 0.0f;
            acc += m[j] * fv;
        }
        out[(((long long)(b * 2 + c) * HO) + (h * up + u)) * (long long)WO + (w * up + v)] = acc * inv;
    }
}

// ---------------------------------------------------------------------------
// Host orchestration
// ---------------------------------------------------------------------------
struct BN { const float *g = nullptr, *b = nullptr, *m = nullptr, *v = nullptr; };
static inline BN bn_of(void* const* d_in, int base) {
    return { (const float*)d_in[base + 1], (const float*)d_in[base + 0],
             (const float*)d_in[base + 2], (const float*)d_in[base + 3] };
}
static inline unsigned gdiv(long long a, long long b) { return (unsigned)((a + b - 1) / b); }

static void launch_gemm16(hipStream_t st, const _Float16* A, const _Float16* Bw, void* OUT,
                          const float* bias, BN bn, int O, int Opad, int Kpad,
                          int H, int W, long long aBS, long long oBS, int ldo,
                          int nb, int flags)
{
    dim3 grid((unsigned)((H * W) / 32), gdiv(Opad, 128), (unsigned)nb);
    gemm16_kernel<<<grid, dim3(128), 0, st>>>(A, Bw, OUT, bias, bn.g, bn.b, bn.m, bn.v,
                                              O, Opad, Kpad, H, W, aBS, oBS, ldo, flags);
}

static void run_dsconv(hipStream_t st, void* const* d_in, const _Float16* src, _Float16* dwtmp,
                       _Float16* dst, const _Float16* w16, int bnBase, int dwIdx,
                       int Cin, int Kpad, int Cout, int B, int H, int W)
{
    int S = H * W;
    long long tot = (long long)B * S * Kpad;
    dwconv_nhwc_kernel<<<gdiv(tot, 256), 256, 0, st>>>(src, (const float*)d_in[dwIdx],
                                                       dwtmp, B, Cin, Kpad, H, W);
    launch_gemm16(st, dwtmp, w16, dst, nullptr, bn_of(d_in, bnBase), Cout, Cout, Kpad,
                  H, W, (long long)S * Kpad, (long long)S * Cout, Cout, B, EF_BN | EF_SILU);
}

static void run_se(hipStream_t st, void* const* d_in, _Float16* x, float* sesc,
                   int b1I, int b2I, int w1I, int w2I, int B, int S)
{
    se_kernel<<<(unsigned)B, 256, 0, st>>>(x, (const float*)d_in[w1I], (const float*)d_in[b1I],
                                           (const float*)d_in[w2I], (const float*)d_in[b2I],
                                           sesc, S, 256, 32);
    long long tot = (long long)B * S * 256;
    se_scale_kernel<<<gdiv(tot, 256), 256, 0, st>>>(x, sesc, 256, S, tot);
}

// block flat layout: +0 ds1.bn, +4 ds1.dw, +5 ds1.pw, +6 ds2.bn, +10 ds2.dw,
//                    +11 ds2.pw, +12 se.b1, +13 se.b2, +14 se.w1, +15 se.w2
static void run_block(hipStream_t st, void* const* d_in, _Float16* x, _Float16* t1, _Float16* t2,
                      float* sesc, const _Float16* wA, const _Float16* wB,
                      int base, int B, int H, int W)
{
    run_dsconv(st, d_in, x, t1, t2, wA, base + 0, base + 4, 256, 256, 256, B, H, W);
    run_se(st, d_in, t2, sesc, base + 12, base + 13, base + 14, base + 15, B, H * W);
    run_dsconv(st, d_in, t2, t1, x, wB, base + 6, base + 10, 256, 256, 256, B, H, W);
}

// Flat input order (JAX pytree: dicts by sorted key, lists in order):
// 0 feat1, 1 feat2, 2-5 bn1, 6-9 bn2, 10 flow_b, 11-14 flow_ds.bn, 15 flow_ds.dw,
// 16 flow_ds.pw, 17 flow_w, 18-21 fuse_in_ds.bn, 22 fuse_in_ds.dw, 23 fuse_in_ds.pw,
// 24-27 fuse_se(b1,b2,w1,w2), 28 proj1, 29 proj2, 30 refine_b, 31-46 refine_block[0],
// 47-50 refine_ds.bn, 51 refine_ds.dw, 52 refine_ds.pw, 53 refine_w,
// 54-69 trunk[0], 70-85 trunk[1], 86 up.b1, 87 up.b2, 88 up.w1, 89 up.w2
extern "C" void kernel_launch(void* const* d_in, const int* in_sizes, int n_in,
                              void* d_out, int out_size, void* d_ws, size_t ws_size,
                              hipStream_t stream)
{
    (void)in_sizes; (void)n_in; (void)out_size; (void)ws_size;
    const int B = 8, Cin = 384, H = 40, W = 40, S = H * W;
    const int P = 128, Fc = 256, CH0 = 465, K0 = 480, KR = 288, UP = 16;
    const int NMASK = UP * UP * 9;                   // 2304
    const int SP = (H + 2) * (W + 2);                // 1764
    auto in = [&](int i) { return (const float*)d_in[i]; };

    // ---- workspace carve-out (256B aligned) ----
    char* base = (char*)d_ws; size_t off = 0;
    auto alloc = [&](size_t bytes) -> char* {
        char* p = base + off; off += ((bytes + 255) / 256) * 256; return p;
    };
    // packed f16 weights
    size_t woff = 0;
    auto walloc = [&](size_t halfs) { size_t o = woff; woff += halfs; return o; };
    size_t WP1 = walloc(128 * 384),  WP2 = walloc(128 * 384);
    size_t WFUSE = walloc((size_t)256 * K0);
    size_t WT0A = walloc(256 * 256), WT0B = walloc(256 * 256);
    size_t WT1A = walloc(256 * 256), WT1B = walloc(256 * 256);
    size_t WFLOWDS = walloc(256 * 256), WRBA = walloc(256 * 256), WRBB = walloc(256 * 256);
    size_t WREFDS = walloc((size_t)256 * KR);
    size_t WFLOWW = walloc(32 * 256), WREFW = walloc(32 * 256);  // Opad=32 for 32-wide N tile
    size_t WUP1 = walloc((size_t)9 * 256 * 256);
    size_t WUP2 = walloc((size_t)NMASK * 256);
    _Float16* WB = (_Float16*)alloc(woff * 2);

    // region R: feat16 (2x [B,S,384]) reused later as X,T1,T2 ([B,S,256] each)
    char* R = alloc((size_t)2 * B * S * 384 * 2);
    _Float16* FEAT1h = (_Float16*)R;
    _Float16* FEAT2h = FEAT1h + (size_t)B * S * 384;
    _Float16* X  = (_Float16*)R;
    _Float16* T1 = X + (size_t)B * S * 256;
    _Float16* T2 = T1 + (size_t)B * S * 256;

    _Float16* F1h  = (_Float16*)alloc((size_t)B * S * P * 2);
    _Float16* F2h  = (_Float16*)alloc((size_t)B * S * P * 2);
    _Float16* X0   = (_Float16*)alloc((size_t)B * S * K0 * 2);  // also reused [B,S,288]/[B,S,256]
    _Float16* T480 = (_Float16*)alloc((size_t)B * S * K0 * 2);  // also reused as T288
    _Float16* XPAD = (_Float16*)alloc((size_t)B * SP * 256 * 2);
    float* FLOW  = (float*)alloc((size_t)B * 2 * S * 4);
    float* SESC  = (float*)alloc((size_t)B * 256 * 4);
    float* MASKB = (float*)alloc((size_t)S * NMASK * 4);

    // ---- weight packing (fp32 -> fp16 [Opad,Kpad], zero padded) ----
    auto wprep = [&](int srcIdx, size_t dstOff, int O, int K, int Opad, int Kpad,
                     int sM, int sK, int elemOff) {
        long long tot = (long long)Opad * Kpad;
        wprep_kernel<<<gdiv(tot, 256), 256, 0, stream>>>(in(srcIdx) + elemOff, WB + dstOff,
                                                         O, K, Kpad, tot, sM, sK);
    };
    wprep(28, WP1, 128, 384, 128, 384, 384, 1, 0);
    wprep(29, WP2, 128, 384, 128, 384, 384, 1, 0);
    wprep(23, WFUSE, 256, CH0, 256, K0, CH0, 1, 0);
    wprep(59, WT0A, 256, 256, 256, 256, 256, 1, 0);
    wprep(65, WT0B, 256, 256, 256, 256, 256, 1, 0);
    wprep(75, WT1A, 256, 256, 256, 256, 256, 1, 0);
    wprep(81, WT1B, 256, 256, 256, 256, 256, 1, 0);
    wprep(16, WFLOWDS, 256, 256, 256, 256, 256, 1, 0);
    wprep(36, WRBA, 256, 256, 256, 256, 256, 1, 0);
    wprep(42, WRBB, 256, 256, 256, 256, 256, 1, 0);
    wprep(52, WREFDS, 256, 258, 256, KR, 258, 1, 0);
    wprep(17, WFLOWW, 2, 256, 32, 256, 256, 1, 0);
    wprep(53, WREFW, 2, 256, 32, 256, 256, 1, 0);
    for (int t = 0; t < 9; ++t)   // up.w1 (O,K,3,3): tap t matrix, elem stride 9
        wprep(88, WUP1 + (size_t)t * 256 * 256, 256, 256, 256, 256, 256 * 9, 9, t);
    wprep(89, WUP2, NMASK, 256, NMASK, 256, 256, 1, 0);

    // ---- feat -> NHWC f16 ----
    { long long tot = (long long)B * S * Cin;
      nchw2nhwc_kernel<<<gdiv(tot, 256), 256, 0, stream>>>(in(0), FEAT1h, B, Cin, S);
      nchw2nhwc_kernel<<<gdiv(tot, 256), 256, 0, stream>>>(in(1), FEAT2h, B, Cin, S); }

    // 1-2) proj + BN + SiLU  (WMMA)
    launch_gemm16(stream, FEAT1h, WB + WP1, F1h, nullptr, bn_of(d_in, 2), P, P, Cin,
                  H, W, (long long)S * Cin, (long long)S * P, P, B, EF_BN | EF_SILU);
    launch_gemm16(stream, FEAT2h, WB + WP2, F2h, nullptr, bn_of(d_in, 6), P, P, Cin,
                  H, W, (long long)S * Cin, (long long)S * P, P, B, EF_BN | EF_SILU);

    // 3-4) x0 = concat([f1, f2, |f1-f2|, corr]) NHWC [B,S,480]
    { long long tot = (long long)B * S * K0;
      concat3_nhwc_kernel<<<gdiv(tot, 256), 256, 0, stream>>>(F1h, F2h, X0, B, P, S, K0, 3 * P, CH0); }
    { long long tot = (long long)B * 81 * S;
      corr_nhwc_kernel<<<gdiv(tot, 256), 256, 0, stream>>>(F1h, F2h, X0, B, P, H, W, K0, 3 * P); }

    // 5-6) fuse dsconv (465->256) + SE
    run_dsconv(stream, d_in, X0, T480, X, WB + WFUSE, 18, 22, CH0, K0, Fc, B, H, W);
    run_se(stream, d_in, X, SESC, 24, 25, 26, 27, B, S);

    // 7) trunk x2
    run_block(stream, d_in, X, T1, T2, SESC, WB + WT0A, WB + WT0B, 54, B, H, W);
    run_block(stream, d_in, X, T1, T2, SESC, WB + WT1A, WB + WT1B, 70, B, H, W);

    // 8) flow = pw(flow_ds(x), flow_w) + flow_b   (f32 CHW)
    run_dsconv(stream, d_in, X, T1, T2, WB + WFLOWDS, 11, 15, Fc, Fc, Fc, B, H, W);
    launch_gemm16(stream, T2, WB + WFLOWW, FLOW, in(10), BN{}, 2, 32, Fc, H, W,
                  (long long)S * Fc, (long long)2 * S, 0, B, EF_BIAS | EF_OUTF32 | EF_CHW);

    // 9) refine_ds(concat([x, flow]))  258->256 (padded to 288)
    { long long tot = (long long)B * S * KR;
      concat_xf_kernel<<<gdiv(tot, 256), 256, 0, stream>>>(X, FLOW, T480, B, Fc, KR, S); }
    run_dsconv(stream, d_in, T480, X0, T1, WB + WREFDS, 47, 51, Fc + 2, KR, Fc, B, H, W);

    // 10) refine block (r in T1; scratch T2 + X0-region)
    run_block(stream, d_in, T1, T2, (_Float16*)X0, SESC, WB + WRBA, WB + WRBB, 31, B, H, W);

    // 11) flow += pw(r, refine_w) + refine_b  (accumulate into f32 CHW FLOW)
    launch_gemm16(stream, T1, WB + WREFW, FLOW, in(30), BN{}, 2, 32, Fc, H, W,
                  (long long)S * Fc, (long long)2 * S, 0, B,
                  EF_BIAS | EF_OUTF32 | EF_CHW | EF_ACC);

    // 12a) mask1 = relu(conv3x3(x, up.w1)+b1): one kernel, 9 taps fused in-register
    { long long tot = (long long)B * SP * 256;
      padcpy_kernel<<<gdiv(tot, 256), 256, 0, stream>>>(X, XPAD, B, Fc, H, W); }
    launch_gemm16(stream, XPAD, WB + WUP1, T2, in(86), BN{}, Fc, Fc, Fc, H, W,
                  (long long)SP * Fc, (long long)S * Fc, Fc, B,
                  EF_CONV3 | EF_BIAS | EF_RELU);

    // 12b) per batch: mask = pw(mask1, up.w2)+b2 (WMMA, O=2304, f32 NHWC) -> upsample
    for (int b = 0; b < B; ++b) {
        launch_gemm16(stream, T2 + (size_t)b * S * Fc, WB + WUP2, MASKB, in(87), BN{},
                      NMASK, NMASK, Fc, H, W, 0, 0, NMASK, 1, EF_BIAS | EF_OUTF32);
        long long tot = (long long)S * UP * UP;
        upsample_kernel<<<gdiv(tot, 256), 256, 0, stream>>>(MASKB, FLOW + (long long)b * 2 * S,
                                                            (float*)d_out, b, H, W, UP);
    }
}
